// MultiHeadAttention_11914239279739
// MI455X (gfx1250) — matrix operand order repaired and checked by reading, not compiled, not run on hardware
//
#include <hip/hip_runtime.h>

// ---------------------------------------------------------------------------
// MI455X (gfx1250) GQA attention: bf16 WMMA everywhere, f32 accumulate.
//   N_HEAD=32, N_KV_HEAD=8, HEAD_DIM=64, N_EMBD=2048, B=2, T=2048
// ---------------------------------------------------------------------------

typedef __attribute__((ext_vector_type(16))) __bf16    v16bf;
typedef __attribute__((ext_vector_type(8)))  float     v8f;
typedef __attribute__((ext_vector_type(4)))  unsigned  v4u;

union Frag { v4u q[2]; v16bf v; };   // 32B = 8 VGPRs = one WMMA A/B operand

__device__ __forceinline__ unsigned short f2bf(float f) {
  union { float f; unsigned u; } v; v.f = f;
  unsigned r = (v.u + 0x7FFFu + ((v.u >> 16) & 1u)) >> 16;   // RNE
  return (unsigned short)r;
}
__device__ __forceinline__ float bf2f(unsigned short b) {
  union { unsigned u; float f; } v; v.u = ((unsigned)b) << 16;
  return v.f;
}

__device__ __forceinline__ v8f wmma_bf16(v16bf a, v16bf b, v8f c) {
  // D = A(16x32 bf16) * B(32x16 bf16) + C(16x16 f32)
  return __builtin_amdgcn_wmma_f32_16x16x32_bf16(false, a, false, b,
                                                 (short)0, c, false, false);
}

// A-fragment (16x32, M x K): lane(0..15)=row r, lanes 16..31 same rows.
// VGPR0-3: K = 8h..8h+7 ; VGPR4-7: K = 16+8h..16+8h+7  (h = lane>>4)
__device__ __forceinline__ v16bf load_a16x32(const unsigned short* __restrict__ base,
                                             int ldr, int lane) {
  int r = lane & 15, h = lane >> 4;
  const unsigned short* row = base + (size_t)r * ldr;
  Frag f;
  f.q[0] = *(const v4u*)(row + 8 * h);        // 16B: K = 8h .. 8h+7
  f.q[1] = *(const v4u*)(row + 16 + 8 * h);   // 16B: K = 16+8h .. 16+8h+7
  return f.v;
}

// B-fragment (32x16, K x N) where column n == row n of a row-major source:
// lane n -> col n; VGPR0-3: K = 8h..8h+7 ; VGPR4-7: K = 16+8h..16+8h+7, 2 vals/VGPR.
__device__ __forceinline__ v16bf load_b32x16(const unsigned short* __restrict__ base,
                                             int ldr, int lane) {
  int n = lane & 15, h = lane >> 4;
  const unsigned short* row = base + (size_t)n * ldr + 8 * h;
  Frag f;
  f.q[0] = *(const v4u*)(row);       // K = 8h .. 8h+7
  f.q[1] = *(const v4u*)(row + 16);  // K = 16+8h .. 16+8h+7
  return f.v;
}

// ---------------------------------------------------------------------------
// fp32 -> bf16 convert
__global__ void k_f32_to_bf16(const float* __restrict__ in,
                              unsigned short* __restrict__ out, int n) {
  int i = blockIdx.x * blockDim.x + threadIdx.x;
  if (i < n) out[i] = f2bf(in[i]);
}

// Build transposed bf16 weights: wqkvT[3072][2048], woT[2048][2048]
__global__ void k_build_w(const float* __restrict__ wq, const float* __restrict__ wk,
                          const float* __restrict__ wv, const float* __restrict__ wo,
                          unsigned short* __restrict__ wqkvT,
                          unsigned short* __restrict__ woT) {
  int idx = blockIdx.x * blockDim.x + threadIdx.x;
  const int NQKV = 3072 * 2048;
  if (idx < NQKV) {
    int n = idx >> 11, k = idx & 2047;
    float v;
    if (n < 2048)      v = wq[(size_t)k * 2048 + n];
    else if (n < 2560) v = wk[(size_t)k * 512 + (n - 2048)];
    else               v = wv[(size_t)k * 512 + (n - 2560)];
    wqkvT[idx] = f2bf(v);
  } else {
    int i2 = idx - NQKV;
    if (i2 < 2048 * 2048) {
      int n = i2 >> 11, k = i2 & 2047;
      woT[i2] = f2bf(wo[(size_t)k * 2048 + n]);
    }
  }
}

// ---------------------------------------------------------------------------
// One K=32 slice of a 32x64 wave tile: 2 A fragments + 4 B fragments.
struct Tile { v16bf a0, a1, b0, b1, b2, b3; };

__device__ __forceinline__ void load_tile(Tile& t,
    const unsigned short* __restrict__ A0, const unsigned short* __restrict__ A1,
    const unsigned short* __restrict__ B0, size_t bs, int K, int k, int lane) {
  t.a0 = load_a16x32(A0 + k, K, lane);
  t.a1 = load_a16x32(A1 + k, K, lane);
  t.b0 = load_b32x16(B0 + k,          K, lane);
  t.b1 = load_b32x16(B0 + bs + k,     K, lane);
  t.b2 = load_b32x16(B0 + 2 * bs + k, K, lane);
  t.b3 = load_b32x16(B0 + 3 * bs + k, K, lane);
}

__device__ __forceinline__ void mma_tile(v8f acc[2][4], const Tile& t) {
  acc[0][0] = wmma_bf16(t.a0, t.b0, acc[0][0]);
  acc[0][1] = wmma_bf16(t.a0, t.b1, acc[0][1]);
  acc[0][2] = wmma_bf16(t.a0, t.b2, acc[0][2]);
  acc[0][3] = wmma_bf16(t.a0, t.b3, acc[0][3]);
  acc[1][0] = wmma_bf16(t.a1, t.b0, acc[1][0]);
  acc[1][1] = wmma_bf16(t.a1, t.b1, acc[1][1]);
  acc[1][2] = wmma_bf16(t.a1, t.b2, acc[1][2]);
  acc[1][3] = wmma_bf16(t.a1, t.b3, acc[1][3]);
}

// WMMA GEMM: C[M x N] = A[M x K](bf16,row-major) * Bt[N x K](bf16, B^T row-major)
// One wave per 32x64 C tile, 8 WMMAs per K=32 step. Ping-pong double buffer:
// each tile buffer is refilled immediately AFTER its last use, so the refill
// loads overlap the other buffer's 8 WMMAs and no register copies are needed
// at the loop back-edge. Speculative tail refills wrap to k=0 (stay in-bounds).
__global__ __launch_bounds__(32) void k_wmma_gemm(
    const unsigned short* __restrict__ A, const unsigned short* __restrict__ Bt,
    unsigned short* __restrict__ Cbf, float* __restrict__ Cf, int K, int ldc) {
  int lane = threadIdx.x;
  int m0 = blockIdx.x * 32;
  int n0 = blockIdx.y * 64;
  v8f acc[2][4];
#pragma unroll
  for (int i = 0; i < 2; ++i)
#pragma unroll
    for (int j = 0; j < 4; ++j) acc[i][j] = (v8f){0.f,0.f,0.f,0.f,0.f,0.f,0.f,0.f};

  const unsigned short* arow0 = A + (size_t)m0 * K;
  const unsigned short* arow1 = A + (size_t)(m0 + 16) * K;
  const unsigned short* b0    = Bt + (size_t)n0 * K;
  const size_t bs = (size_t)16 * K;

  Tile t0, t1;
  load_tile(t0, arow0, arow1, b0, bs, K, 0, lane);
  load_tile(t1, arow0, arow1, b0, bs, K, 32, lane);

#pragma unroll 1
  for (int k0 = 0; k0 < K; k0 += 64) {       // K is a multiple of 64
    mma_tile(acc, t0);
    load_tile(t0, arow0, arow1, b0, bs, K, (k0 + 64 < K) ? k0 + 64 : 0, lane);
    mma_tile(acc, t1);
    load_tile(t1, arow0, arow1, b0, bs, K, (k0 + 96 < K) ? k0 + 96 : 0, lane);
  }

  int nn = lane & 15, hh = lane >> 4;
#pragma unroll
  for (int i = 0; i < 2; ++i)
#pragma unroll
    for (int j = 0; j < 4; ++j)
#pragma unroll
      for (int v = 0; v < 8; ++v) {
        int row = m0 + 16 * i + v + 8 * hh;
        int col = n0 + 16 * j + nn;
        if (Cbf) Cbf[(size_t)row * ldc + col] = f2bf(acc[i][j][v]);
        else     Cf [(size_t)row * ldc + col] = acc[i][j][v];
      }
}

// ---------------------------------------------------------------------------
// RoPE in-place on bf16 qkv [4096 x 3072]; folds 1/sqrt(64) into Q.
__global__ void k_rope(unsigned short* __restrict__ qkv) {
  int idx = blockIdx.x * blockDim.x + threadIdx.x;
  const int total = 2 * 2048 * 40 * 32;       // B*T*(32 Q + 8 KV heads)*32 pairs
  if (idx >= total) return;
  int i = idx & 31;
  int r = idx >> 5;
  int head = r % 40;
  int bt = r / 40;                             // b*T + t
  int t = bt & 2047;
  int col = (head < 32) ? head * 64 : 2048 + (head - 32) * 64;
  unsigned short* p = qkv + (size_t)bt * 3072 + col;
  // freq = 10000^(-2i/64) = exp(-(i/32)*ln(1e4))
  float inv = __expf(-(float)i * (9.210340371976184f / 32.0f));
  float theta = (float)t * inv;
  float s, c;
  __sincosf(theta, &s, &c);
  float v0 = bf2f(p[i]);
  float v1 = bf2f(p[i + 32]);
  float o0 = v0 * c - v1 * s;
  float o1 = v1 * c + v0 * s;
  if (head < 32) { o0 *= 0.125f; o1 *= 0.125f; }   // attention scale into Q
  p[i]      = f2bf(o0);
  p[i + 32] = f2bf(o1);
}

// V -> Vt [B][8][64][T] so PV B-fragments are contiguous along key positions.
__global__ void k_vtrans(const unsigned short* __restrict__ qkv,
                         unsigned short* __restrict__ vt) {
  int idx = blockIdx.x * blockDim.x + threadIdx.x;
  if (idx >= 2 * 8 * 64 * 2048) return;
  int t = idx & 2047;
  int r = idx >> 11;
  int d = r & 63;
  int r2 = r >> 6;
  int kvh = r2 & 7;
  int b = r2 >> 3;
  vt[idx] = qkv[(size_t)(b * 2048 + t) * 3072 + 2560 + kvh * 64 + d];
}

// ---------------------------------------------------------------------------
// Flash attention: one wave per (b, h, 16-query tile). Per 32-key block:
// prefetch next block (global_prefetch_b8), load all 8 fragments (4 K, 4 V)
// up front so V-load latency hides under the 4 QK^T WMMAs and the softmax
// VALU work, causal mask via selects (EXEC stays all-ones for WMMA), online
// softmax with shfl_xor row butterflies, P through LDS (C->A layout), 4 PV
// WMMAs into a 16x64 f32 accumulator.
__global__ __launch_bounds__(32) void k_attn(
    const unsigned short* __restrict__ qkv,
    const unsigned short* __restrict__ vt,
    unsigned short* __restrict__ attn_out) {
  __shared__ __align__(16) unsigned short plds[16 * 32];
  const int T = 2048, LDQ = 3072;
  int lane = threadIdx.x;
  int q0 = blockIdx.x * 16;
  int h  = blockIdx.y;
  int b  = blockIdx.z;
  int kvh = h >> 2;                            // repeat_interleave(4)
  int nn = lane & 15, hh = lane >> 4;

  const unsigned short* qbase = qkv + (size_t)(b * T + q0) * LDQ + h * 64;
  v16bf a0 = load_a16x32(qbase,      LDQ, lane);   // d 0..31
  v16bf a1 = load_a16x32(qbase + 32, LDQ, lane);   // d 32..63

  float m[8], l[8];
  v8f o[4];
#pragma unroll
  for (int v = 0; v < 8; ++v) { m[v] = -3.0e38f; l[v] = 0.f; }
#pragma unroll
  for (int j = 0; j < 4; ++j) o[j] = (v8f){0.f,0.f,0.f,0.f,0.f,0.f,0.f,0.f};

  const unsigned short* kb_base = qkv + (size_t)(b * T) * LDQ + 2048 + kvh * 64;
  const unsigned short* v_base  = vt + (size_t)((b * 8 + kvh) * 64) * T;
  const v8f zero = (v8f){0.f,0.f,0.f,0.f,0.f,0.f,0.f,0.f};

  int kend = q0 + 16;                          // causal: keys 0 .. q0+15
  for (int kb = 0; kb < kend; kb += 32) {
    const unsigned short* kp = kb_base + (size_t)kb * LDQ;

    // Prefetch next key block: 32 K rows (one per lane) + 64 Vt line segments.
    // Runs one block ahead; stays inside the mapped workspace on the last step.
    __builtin_prefetch(kp + (size_t)(32 + lane) * LDQ, 0, 3);
    __builtin_prefetch(v_base + (size_t)lane * T        + kb + 32, 0, 3);
    __builtin_prefetch(v_base + (size_t)(lane + 32) * T + kb + 32, 0, 3);

    // All fragment loads for this block issued up front (one wait, 8 WMMAs).
    v16bf bk0 = load_b32x16(kp,                         LDQ, lane);
    v16bf bk1 = load_b32x16(kp + 32,                    LDQ, lane);
    v16bf bk2 = load_b32x16(kp + (size_t)16 * LDQ,      LDQ, lane);
    v16bf bk3 = load_b32x16(kp + (size_t)16 * LDQ + 32, LDQ, lane);
    v16bf bv0 = load_b32x16(v_base + (size_t)( 0) * T + kb, T, lane);
    v16bf bv1 = load_b32x16(v_base + (size_t)(16) * T + kb, T, lane);
    v16bf bv2 = load_b32x16(v_base + (size_t)(32) * T + kb, T, lane);
    v16bf bv3 = load_b32x16(v_base + (size_t)(48) * T + kb, T, lane);

    v8f s0 = zero, s1 = zero;
    s0 = wmma_bf16(a0, bk0, s0);
    s0 = wmma_bf16(a1, bk1, s0);
    s1 = wmma_bf16(a0, bk2, s1);
    s1 = wmma_bf16(a1, bk3, s1);

    float alpha[8];
#pragma unroll
    for (int v = 0; v < 8; ++v) {
      int row = q0 + v + 8 * hh;
      float x0 = (kb + nn      <= row) ? s0[v] : -3.0e38f;
      float x1 = (kb + 16 + nn <= row) ? s1[v] : -3.0e38f;
      float mx = fmaxf(x0, x1);                       // row reduce (16-lane group)
      mx = fmaxf(mx, __shfl_xor(mx, 1, 32));
      mx = fmaxf(mx, __shfl_xor(mx, 2, 32));
      mx = fmaxf(mx, __shfl_xor(mx, 4, 32));
      mx = fmaxf(mx, __shfl_xor(mx, 8, 32));
      float mn = fmaxf(m[v], mx);
      float al = __expf(m[v] - mn);
      float e0 = __expf(x0 - mn);
      float e1 = __expf(x1 - mn);
      float rs = e0 + e1;
      rs += __shfl_xor(rs, 1, 32);
      rs += __shfl_xor(rs, 2, 32);
      rs += __shfl_xor(rs, 4, 32);
      rs += __shfl_xor(rs, 8, 32);
      l[v] = l[v] * al + rs;
      m[v] = mn;
      alpha[v] = al;
      plds[(v + 8 * hh) * 32 + nn]      = f2bf(e0);   // P tile, row-major 16x32
      plds[(v + 8 * hh) * 32 + 16 + nn] = f2bf(e1);
    }
#pragma unroll
    for (int j = 0; j < 4; ++j)
#pragma unroll
      for (int v = 0; v < 8; ++v) o[j][v] *= alpha[v];

    __syncthreads();
    // Reload P as a WMMA A-fragment (C-layout -> A-layout reshape via LDS)
    Frag pf;
    const unsigned* l32 = (const unsigned*)plds;
    pf.q[0] = *(const v4u*)(l32 + nn * 16 + 4 * hh);       // K = 8h..8h+7
    pf.q[1] = *(const v4u*)(l32 + nn * 16 + 8 + 4 * hh);   // K = 16+8h..
    o[0] = wmma_bf16(pf.v, bv0, o[0]);
    o[1] = wmma_bf16(pf.v, bv1, o[1]);
    o[2] = wmma_bf16(pf.v, bv2, o[2]);
    o[3] = wmma_bf16(pf.v, bv3, o[3]);
    __syncthreads();
  }

#pragma unroll
  for (int j = 0; j < 4; ++j)
#pragma unroll
    for (int v = 0; v < 8; ++v) {
      int row = q0 + v + 8 * hh;
      int col = h * 64 + 16 * j + nn;
      attn_out[(size_t)(b * T + row) * 2048 + col] = f2bf(o[j][v] / l[v]);
    }
}

// ---------------------------------------------------------------------------
extern "C" void kernel_launch(void* const* d_in, const int* in_sizes, int n_in,
                              void* d_out, int out_size, void* d_ws, size_t ws_size,
                              hipStream_t stream) {
  (void)in_sizes; (void)n_in; (void)out_size; (void)ws_size;
  const float* x  = (const float*)d_in[0];
  const float* wq = (const float*)d_in[1];
  const float* wk = (const float*)d_in[2];
  const float* wv = (const float*)d_in[3];
  const float* wo = (const float*)d_in[4];
  float* out = (float*)d_out;

  unsigned short* ws = (unsigned short*)d_ws;
  const size_t N_X    = 4096ull * 2048;   // x bf16
  const size_t N_WQKV = 3072ull * 2048;   // [wq|wk|wv]^T bf16
  const size_t N_WO   = 2048ull * 2048;   // wo^T bf16
  const size_t N_QKV  = 4096ull * 3072;   // fused qkv bf16
  const size_t N_VT   = 2ull * 8 * 64 * 2048;
  unsigned short* xbf   = ws;
  unsigned short* wqkvT = xbf + N_X;
  unsigned short* woT   = wqkvT + N_WQKV;
  unsigned short* qkv   = woT + N_WO;
  unsigned short* vt    = qkv + N_QKV;
  unsigned short* aout  = vt + N_VT;      // attention output bf16 [4096 x 2048]

  k_f32_to_bf16<<<(int)((N_X + 255) / 256), 256, 0, stream>>>(x, xbf, (int)N_X);

  const int NW = 3072 * 2048 + 2048 * 2048;
  k_build_w<<<(NW + 255) / 256, 256, 0, stream>>>(wq, wk, wv, wo, wqkvT, woT);

  // QKV = x @ [wq|wk|wv]   (M=4096, N=3072, K=2048), 32x64 tile per wave
  k_wmma_gemm<<<dim3(4096 / 32, 3072 / 64), 32, 0, stream>>>(
      xbf, wqkvT, qkv, nullptr, 2048, 3072);

  k_rope<<<(2 * 2048 * 40 * 32 + 255) / 256, 256, 0, stream>>>(qkv);
  k_vtrans<<<(2 * 8 * 64 * 2048 + 255) / 256, 256, 0, stream>>>(qkv, vt);

  // Flash attention: 128 q-tiles x 32 heads x 2 batches, 1 wave each
  k_attn<<<dim3(2048 / 16, 32, 2), 32, 0, stream>>>(qkv, vt, aout);

  // out = attn @ wo        (M=4096, N=2048, K=2048), fp32 store
  k_wmma_gemm<<<dim3(4096 / 32, 2048 / 64), 32, 0, stream>>>(
      aout, woT, nullptr, out, 2048, 2048);
}
